// CAISFormerBlock_10307921511023
// MI455X (gfx1250) — matrix operand order repaired and checked by reading, not compiled, not run on hardware
//
#include <hip/hip_runtime.h>

typedef __attribute__((ext_vector_type(16))) _Float16 v16h;
typedef __attribute__((ext_vector_type(8)))  _Float16 v8h;
typedef __attribute__((ext_vector_type(8)))  float    v8f;

static constexpr int CB  = 2;       // batch
static constexpr int CC  = 64;      // channels
static constexpr int CH  = 64;
static constexpr int CWd = 64;
static constexpr int CN  = CH * CWd;   // 4096 spatial
static constexpr int CNB = CB * CN;    // 8192

#define DEV __device__ __forceinline__

// A-fragment (16x32, 16-bit) K offset for element i in lane (ISA 7.12.2)
DEV int koffA(int i, int lane) {
    return (i < 8 ? i : i + 8) + ((lane & 16) ? 8 : 0);
}

DEV v8f wmma_f16(v16h a, v16h b, v8f c) {
    return __builtin_amdgcn_wmma_f32_16x16x32_f16(false, a, false, b, (short)0, c, false, false);
}

DEV v16h cat8(v8h a, v8h b) {
    return __builtin_shufflevector(a, b, 0, 1, 2, 3, 4, 5, 6, 7, 8, 9, 10, 11, 12, 13, 14, 15);
}

// ---- 16-lane butterfly reductions via DPP16 (pure VALU, no LDS pipe) ------
// quad_perm(1,0,3,2)=0xB1 (xor1), quad_perm(2,3,0,1)=0x4E (xor2),
// ROW_HALF_MIRROR=0x141 (xor4 once quads uniform), ROW_MIRROR=0x140 (xor8).
template <int CTRL>
DEV float dpp_bcast(float v) {
    return __builtin_bit_cast(float,
        __builtin_amdgcn_update_dpp(0, __builtin_bit_cast(int, v), CTRL, 0xF, 0xF, true));
}
DEV float red16_max(float v) {
    v = fmaxf(v, dpp_bcast<0xB1>(v));
    v = fmaxf(v, dpp_bcast<0x4E>(v));
    v = fmaxf(v, dpp_bcast<0x141>(v));
    v = fmaxf(v, dpp_bcast<0x140>(v));
    return v;
}
DEV float red16_sum(float v) {
    v += dpp_bcast<0xB1>(v);
    v += dpp_bcast<0x4E>(v);
    v += dpp_bcast<0x141>(v);
    v += dpp_bcast<0x140>(v);
    return v;
}

DEV float gelu_exact(float x) { return 0.5f * x * (1.0f + erff(x * 0.70710678118654752f)); }

DEV float epilogue(float x, int ep) {
    if (ep == 1) return gelu_exact(x);
    if (ep == 2) return fmaxf(x, 0.f);
    if (ep == 3) return 1.0f / (1.0f + __expf(-x));
    return x;
}

// ---------------------------------------------------------------------------
// 1x1 conv == GEMM  out[b,o,n] = act(sum_i W[o,i] * in[b,i,n] (*in2) + bias[o]) (+res)
// One wave per 16-wide n strip; M tiled over Cout; K over Cin; WMMA f16->f32.
// ---------------------------------------------------------------------------
template <int Cin, int Cout>
__global__ __launch_bounds__(256) void k_conv1x1(
    const float* __restrict__ in, const float* __restrict__ in2,
    const float* __restrict__ wt, const float* __restrict__ bias,
    const float* __restrict__ res, float* __restrict__ out, int ep)
{
    constexpr int KS = (Cin + 31) / 32;
    constexpr int MT = Cout / 16;
    const int lane = threadIdx.x & 31;
    const int wid  = threadIdx.x >> 5;
    const int n0   = (blockIdx.x * 8 + wid) * 16;
    const int b    = n0 >> 12;
    const int nl   = n0 & (CN - 1);
    const int col  = nl + (lane & 15);
    const int hi   = (lane & 16) ? 1 : 0;

    v16h bf[KS];
#pragma unroll
    for (int ks = 0; ks < KS; ++ks) {
        v16h f;
#pragma unroll
        for (int i = 0; i < 16; ++i) {
            int k = ks * 32 + (hi ? 16 : 0) + i;
            float v = 0.f;
            if (k < Cin) {
                size_t idx = ((size_t)b * Cin + k) * CN + col;
                v = in[idx];
                if (in2) v *= in2[idx];
            }
            f[i] = (_Float16)v;
        }
        bf[ks] = f;
    }
#pragma unroll
    for (int mt = 0; mt < MT; ++mt) {
        v8f acc = {};
#pragma unroll
        for (int ks = 0; ks < KS; ++ks) {
            v16h af;
#pragma unroll
            for (int i = 0; i < 16; ++i) {
                int k = ks * 32 + (hi ? 16 : 0) + i;
                int m = mt * 16 + (lane & 15);
                af[i] = (_Float16)((k < Cin) ? wt[(size_t)m * Cin + k] : 0.f);
            }
            acc = wmma_f16(af, bf[ks], acc);
        }
#pragma unroll
        for (int e = 0; e < 8; ++e) {
            int row = mt * 16 + e + hi * 8;
            size_t oidx = ((size_t)b * Cout + row) * CN + col;
            float v = epilogue(acc[e] + bias[row], ep);
            if (res) v += res[oidx];
            out[oidx] = v;
        }
    }
}

// ---------------------------------------------------------------------------
// Flash attention over HW x HW.  out[b,c,n] = sum_m softmax_m(q.k/scale) v[b,c,m]
// 8 waves/block, each wave owns a 16-query tile.  K/V tiles staged in LDS as
// f16 (converted once at stage time), laid out so every WMMA B fragment is a
// pair of 8-half row runs -> two ds_load_b128 per fragment, no per-wave cvt.
// Softmax row reductions run on the VALU via DPP16 butterflies.
// ---------------------------------------------------------------------------
__global__ __launch_bounds__(256) void k_attention(
    const float* __restrict__ q, const float* __restrict__ k,
    const float* __restrict__ v, float* __restrict__ out, float inv_scale)
{
    __shared__ _Float16 ldsK[32][72];      // [m][c]  (transposed), padded rows
    __shared__ _Float16 ldsV[64][40];      // [c][m], padded rows
    __shared__ _Float16 ldsP[8][16][40];   // per-wave P tile [n][m], padded

    const int lane = threadIdx.x & 31;
    const int wid  = threadIdx.x >> 5;
    const int b    = blockIdx.y;
    const int n0   = blockIdx.x * 128 + wid * 16;
    const int hi   = (lane & 16) ? 1 : 0;
    const int l15  = lane & 15;

    // Q^T tile as two A fragments (rows = queries, K-dim = channel 0..63)
    v16h aq0, aq1;
#pragma unroll
    for (int i = 0; i < 16; ++i) {
        int c = hi * 16 + i;
        aq0[i] = (_Float16)q[((size_t)b * CC + c) * CN + n0 + l15];
        aq1[i] = (_Float16)q[((size_t)b * CC + c + 32) * CN + n0 + l15];
    }

    float rmax[8], rsum[8];
    v8f oacc[4];
#pragma unroll
    for (int e = 0; e < 8; ++e) { rmax[e] = -INFINITY; rsum[e] = 0.f; }
#pragma unroll
    for (int ct = 0; ct < 4; ++ct) oacc[ct] = {};

    for (int m0 = 0; m0 < CN; m0 += 32) {
        __syncthreads();
        // stage K (transposed to [m][c]) and V ([c][m]) as f16
        for (int e = threadIdx.x; e < CC * 32; e += 256) {
            int c = e >> 5, mm = e & 31;
            size_t gidx = ((size_t)b * CC + c) * CN + m0 + mm;
            ldsK[mm][c] = (_Float16)k[gidx];
            ldsV[c][mm] = (_Float16)v[gidx];
        }
        __syncthreads();

        // S sub-tiles (16q x 16m) x2, K-dim = 64 channels (2 wmma steps each)
        v8f s0 = {}, s1 = {};
#pragma unroll
        for (int ks = 0; ks < 2; ++ks) {
            int coff = ks * 32 + hi * 16;
            v16h bk0 = cat8(*(const v8h*)&ldsK[l15][coff],
                            *(const v8h*)&ldsK[l15][coff + 8]);
            v16h bk1 = cat8(*(const v8h*)&ldsK[16 + l15][coff],
                            *(const v8h*)&ldsK[16 + l15][coff + 8]);
            s0 = wmma_f16(ks ? aq1 : aq0, bk0, s0);
            s1 = wmma_f16(ks ? aq1 : aq0, bk1, s1);
        }

        // online softmax: rows live in VGPR slot e (+8 for upper half-wave),
        // 16 columns across a half-wave -> DPP16 butterfly reductions
        float p0[8], p1[8], alpha[8];
#pragma unroll
        for (int e = 0; e < 8; ++e) {
            float a0 = s0[e] * inv_scale, a1 = s1[e] * inv_scale;
            float mx = red16_max(fmaxf(a0, a1));
            float mnew = fmaxf(rmax[e], mx);
            alpha[e] = __expf(rmax[e] - mnew);
            rmax[e]  = mnew;
            p0[e] = __expf(a0 - mnew);
            p1[e] = __expf(a1 - mnew);
            float ps = red16_sum(p0[e] + p1[e]);
            rsum[e] = rsum[e] * alpha[e] + ps;
        }
#pragma unroll
        for (int ct = 0; ct < 4; ++ct)
#pragma unroll
            for (int e = 0; e < 8; ++e) oacc[ct][e] *= alpha[e];

        // P (D layout, f32) -> f16 A layout via per-wave LDS scratch
#pragma unroll
        for (int e = 0; e < 8; ++e) {
            int row = e + hi * 8;
            ldsP[wid][row][l15]      = (_Float16)p0[e];
            ldsP[wid][row][16 + l15] = (_Float16)p1[e];
        }
        asm volatile("s_wait_dscnt 0" ::: "memory");
        v16h ap = cat8(*(const v8h*)&ldsP[wid][l15][hi * 8],
                       *(const v8h*)&ldsP[wid][l15][16 + hi * 8]);

        // O += P * V^T  (4 c-tiles of 16; K-dim = 32 keys)
#pragma unroll
        for (int ct = 0; ct < 4; ++ct) {
            int row = ct * 16 + l15;
            v16h bv = cat8(*(const v8h*)&ldsV[row][hi * 8],
                           *(const v8h*)&ldsV[row][16 + hi * 8]);
            oacc[ct] = wmma_f16(ap, bv, oacc[ct]);
        }
    }

#pragma unroll
    for (int ct = 0; ct < 4; ++ct)
#pragma unroll
        for (int e = 0; e < 8; ++e) {
            int c = ct * 16 + l15;
            int n = n0 + e + hi * 8;
            out[((size_t)b * CC + c) * CN + n] = oacc[ct][e] / rsum[e];
        }
}

// ---------------------------------------------------------------------------
// Dense 3x3 conv (clr), LDS-tiled 16x16 spatial tile per block
// ---------------------------------------------------------------------------
__global__ __launch_bounds__(256) void k_conv3x3(
    const float* __restrict__ x, const float* __restrict__ wt,
    const float* __restrict__ bias, float* __restrict__ out)
{
    __shared__ float tile[18 * 18];
    const int tx = threadIdx.x & 15, ty = threadIdx.x >> 4;
    const int ox = blockIdx.x * 16, oy = blockIdx.y * 16;
    const int bz = blockIdx.z;            // b*64 + o
    const int b = bz >> 6, o = bz & 63;
    float acc = 0.f;
    for (int i = 0; i < CC; ++i) {
        const float* xp = x + ((size_t)b * CC + i) * CN;
        for (int e = threadIdx.x; e < 18 * 18; e += 256) {
            int yy = e / 18 + oy - 1, xx = e % 18 + ox - 1;
            tile[e] = (yy >= 0 && yy < CH && xx >= 0 && xx < CWd) ? xp[yy * CWd + xx] : 0.f;
        }
        __syncthreads();
        const float* w9 = wt + ((size_t)o * CC + i) * 9;
#pragma unroll
        for (int kh = 0; kh < 3; ++kh)
#pragma unroll
            for (int kw = 0; kw < 3; ++kw)
                acc = fmaf(w9[kh * 3 + kw], tile[(ty + kh) * 18 + tx + kw], acc);
        __syncthreads();
    }
    out[((size_t)b * CC + o) * CN + (oy + ty) * CWd + ox + tx] = acc + bias[o];
}

// ---------------------------------------------------------------------------
// Depthwise 3x3 (+ optional exact gelu)
// ---------------------------------------------------------------------------
__global__ void k_dw3x3(const float* __restrict__ x, const float* __restrict__ wt,
                        const float* __restrict__ bias, float* __restrict__ out,
                        int Ch, int do_gelu)
{
    size_t idx = (size_t)blockIdx.x * blockDim.x + threadIdx.x;
    if (idx >= (size_t)CB * Ch * CN) return;
    int n = idx & (CN - 1);
    int c = (int)((idx >> 12) % Ch);
    int h = n >> 6, w = n & 63;
    const float* xp = x + (idx - n);
    const float* w9 = wt + c * 9;
    float acc = bias[c];
#pragma unroll
    for (int kh = 0; kh < 3; ++kh) {
        int yy = h + kh - 1;
        if (yy < 0 || yy >= CH) continue;
#pragma unroll
        for (int kw = 0; kw < 3; ++kw) {
            int xx = w + kw - 1;
            if (xx < 0 || xx >= CWd) continue;
            acc = fmaf(w9[kh * 3 + kw], xp[yy * CWd + xx], acc);
        }
    }
    out[idx] = do_gelu ? gelu_exact(acc) : acc;
}

// ---------------------------------------------------------------------------
// LayerNorm over channel dim (per pixel), optional fused LeakyReLU(0.01)
// ---------------------------------------------------------------------------
__global__ void k_layernorm(const float* __restrict__ x, const float* __restrict__ g,
                            const float* __restrict__ bt, float* __restrict__ out, int leaky)
{
    int t = blockIdx.x * blockDim.x + threadIdx.x;  // b*CN + n
    if (t >= CNB) return;
    int b = t >> 12, n = t & (CN - 1);
    const float* xp = x + (size_t)b * CC * CN + n;
    float s = 0.f, s2 = 0.f;
    for (int c = 0; c < CC; ++c) { float vv = xp[c * CN]; s += vv; s2 += vv * vv; }
    float mu  = s * (1.f / CC);
    float var = s2 * (1.f / CC) - mu * mu;
    float rs  = rsqrtf(var + 1e-5f);
    float* op = out + (size_t)b * CC * CN + n;
    for (int c = 0; c < CC; ++c) {
        float y = (xp[c * CN] - mu) * rs * g[c] + bt[c];
        if (leaky && y < 0.f) y *= 0.01f;
        op[c * CN] = y;
    }
}

__global__ void k_meanpool(const float* __restrict__ x, float* __restrict__ gp)
{
    __shared__ float red[256];
    int bc = blockIdx.x;
    const float* xp = x + (size_t)bc * CN;
    float s = 0.f;
    for (int n = threadIdx.x; n < CN; n += 256) s += xp[n];
    red[threadIdx.x] = s;
    __syncthreads();
    for (int st = 128; st > 0; st >>= 1) {
        if ((int)threadIdx.x < st) red[threadIdx.x] += red[threadIdx.x + st];
        __syncthreads();
    }
    if (threadIdx.x == 0) gp[bc] = red[0] * (1.f / CN);
}

// gla channel attention: cw[b,c] = sigmoid(W2 relu(W1 gp + b1) + b2)
__global__ void k_channel_attn(const float* __restrict__ gp,
                               const float* __restrict__ w1, const float* __restrict__ b1,
                               const float* __restrict__ w2, const float* __restrict__ b2,
                               float* __restrict__ cw)
{
    __shared__ float t1[CB * 8];
    int t = threadIdx.x;
    if (t < CB * 8) {
        int b = t >> 3, j = t & 7;
        float s = b1[j];
        for (int c = 0; c < CC; ++c) s += w1[j * CC + c] * gp[b * CC + c];
        t1[t] = fmaxf(s, 0.f);
    }
    __syncthreads();
    if (t < CB * CC) {
        int b = t >> 6, c = t & 63;
        float s = b2[c];
        for (int j = 0; j < 8; ++j) s += w2[c * 8 + j] * t1[b * 8 + j];
        cw[t] = 1.f / (1.f + __expf(-s));
    }
}

__global__ void k_gla_fuse(const float* __restrict__ g, const float* __restrict__ cw,
                           const float* __restrict__ loc, const float* __restrict__ sw,
                           float* __restrict__ out)
{
    size_t idx = (size_t)blockIdx.x * 256 + threadIdx.x;
    if (idx >= (size_t)CB * CC * CN) return;
    int bc = (int)(idx >> 12);
    out[idx] = g[idx] * cw[bc] + loc[idx] * sw[idx];
}

// ---------------------------------------------------------------------------
extern "C" void kernel_launch(void* const* d_in, const int* in_sizes, int n_in,
                              void* d_out, int out_size, void* d_ws, size_t ws_size,
                              hipStream_t stream)
{
    (void)in_sizes; (void)n_in; (void)out_size; (void)ws_size;
    auto IN = [&](int i) { return (const float*)d_in[i]; };

    float* ws = (float*)d_ws;
    size_t off = 0;
    auto alloc = [&](size_t nf) { float* p = ws + off; off += nf; return p; };
    const size_t SZ  = (size_t)CB * CC * CN;   // 64-ch tensor
    const size_t SZ2 = SZ * 2;                 // 128-ch tensor

    float* A0  = alloc(SZ);    // cover state
    float* A1  = alloc(SZ);    // secret state
    float* XN  = alloc(SZ);
    float* XN2 = alloc(SZ);
    float* Qb  = alloc(SZ);
    float* Kb  = alloc(SZ);
    float* Vb  = alloc(SZ);
    float* AT  = alloc(SZ);
    float* T2  = alloc(SZ2);
    float* B1f = alloc(SZ2);
    float* B2f = alloc(SZ2);
    float* LOC = alloc(SZ);
    float* SW  = alloc(SZ);
    float* FU  = alloc(SZ);
    float* GP  = alloc(256);
    float* CWv = alloc(256);

    const dim3 T256(256);
    const dim3 Gconv1(CNB / 16 / 8);           // 64 blocks (8 waves x 16-col strips)

    auto conv1 = [&](const float* in, int Cin, float* out, int Cout,
                     const float* w, const float* b, const float* in2,
                     const float* res, int ep) {
        if (Cin == 64 && Cout == 64)
            k_conv1x1<64, 64><<<Gconv1, T256, 0, stream>>>(in, in2, w, b, res, out, ep);
        else if (Cin == 64 && Cout == 128)
            k_conv1x1<64, 128><<<Gconv1, T256, 0, stream>>>(in, in2, w, b, res, out, ep);
        else if (Cin == 128 && Cout == 64)
            k_conv1x1<128, 64><<<Gconv1, T256, 0, stream>>>(in, in2, w, b, res, out, ep);
        else if (Cin == 64 && Cout == 16)
            k_conv1x1<64, 16><<<Gconv1, T256, 0, stream>>>(in, in2, w, b, res, out, ep);
        else if (Cin == 16 && Cout == 64)
            k_conv1x1<16, 64><<<Gconv1, T256, 0, stream>>>(in, in2, w, b, res, out, ep);
    };
    auto ln = [&](const float* x, const float* g, const float* b, float* out, int leaky) {
        k_layernorm<<<dim3(CNB / 256), T256, 0, stream>>>(x, g, b, out, leaky);
    };
    auto dw = [&](const float* x, const float* w, const float* b, float* out, int Ch, int gel) {
        k_dw3x3<<<dim3((unsigned)((size_t)CB * Ch * CN / 256)), T256, 0, stream>>>(x, w, b, out, Ch, gel);
    };
    auto attend = [&](const float* q, const float* k, const float* v, float* out) {
        k_attention<<<dim3(CN / 128, CB), T256, 0, stream>>>(q, k, v, out, 0.125f);
    };

    // clr: conv3x3 -> LN -> leaky
    auto clr = [&](const float* src, float* dst) {
        k_conv3x3<<<dim3(4, 4, CB * CC), T256, 0, stream>>>(src, IN(2), IN(3), XN);
        ln(XN, IN(4), IN(5), dst, 1);
    };
    // cta: ln -> q/k/v -> attend -> o + residual
    auto cta = [&](float* x) {
        ln(x, IN(6), IN(7), XN, 0);
        conv1(XN, 64, Qb, 64, IN(8), IN(9), nullptr, nullptr, 0);
        dw(XN, IN(10), IN(11), Kb, 64, 0);
        dw(XN, IN(12), IN(13), Vb, 64, 0);
        attend(Qb, Kb, Vb, AT);
        conv1(AT, 64, x, 64, IN(14), IN(15), nullptr, x, 0);
    };
    // nle(x)+x, params start at index pi (ln_w); result -> outp
    auto nle = [&](int pi, float* x, float* outp) {
        ln(x, IN(pi), IN(pi + 1), XN, 0);
        conv1(XN, 64, T2, 128, IN(pi + 2), IN(pi + 3), nullptr, nullptr, 0);
        dw(T2, IN(pi + 4), IN(pi + 5), B1f, 128, 1);
        conv1(XN, 64, T2, 128, IN(pi + 6), IN(pi + 7), nullptr, nullptr, 0);
        dw(T2, IN(pi + 8), IN(pi + 9), B2f, 128, 1);
        conv1(B1f, 128, outp, 64, IN(pi + 10), IN(pi + 11), B2f, x, 0);
    };

    // ---- cover branch ----
    clr(IN(0), A0);
    cta(A0);
    nle(16, A0, A0);
    // ---- secret branch ----
    clr(IN(1), A1);
    cta(A1);
    nle(16, A1, A1);

    // ---- cca: k,v from cover-LN, q from secret-LN ----
    ln(A0, IN(28), IN(29), XN, 0);
    ln(A1, IN(30), IN(31), XN2, 0);
    dw(XN, IN(34), IN(35), Kb, 64, 0);
    dw(XN, IN(36), IN(37), Vb, 64, 0);
    conv1(XN2, 64, Qb, 64, IN(32), IN(33), nullptr, nullptr, 0);
    attend(Qb, Kb, Vb, AT);
    conv1(AT, 64, A0, 64, IN(38), IN(39), nullptr, A0, 0);
    nle(40, A0, A0);

    // ---- gla ----
    conv1(A0, 64, Qb, 64, IN(52), IN(53), nullptr, nullptr, 0);
    dw(A0, IN(54), IN(55), Kb, 64, 0);
    dw(A0, IN(56), IN(57), Vb, 64, 0);
    attend(Qb, Kb, Vb, AT);                               // g
    dw(Vb, IN(58), IN(59), LOC, 64, 0);                   // loc
    conv1(LOC, 64, T2, 16, IN(60), IN(61), nullptr, nullptr, 2);   // sa1 + relu
    conv1(T2, 16, SW, 64, IN(62), IN(63), nullptr, nullptr, 3);    // sa2 + sigmoid
    k_meanpool<<<dim3(CB * CC), T256, 0, stream>>>(AT, GP);
    k_channel_attn<<<dim3(1), dim3(128), 0, stream>>>(GP, IN(64), IN(65), IN(66), IN(67), CWv);
    k_gla_fuse<<<dim3((unsigned)(SZ / 256)), T256, 0, stream>>>(AT, CWv, LOC, SW, FU);
    conv1(FU, 64, A0, 64, IN(68), IN(69), nullptr, A0, 0);

    // ---- final nle, writes d_out directly ----
    nle(70, A0, (float*)d_out);
}
